// ElectraSelfAttentionStochastic_65884798321426
// MI455X (gfx1250) — matrix operand order repaired and checked by reading, not compiled, not run on hardware
//
#include <hip/hip_runtime.h>
#include <hip/hip_bf16.h>
#include <math.h>

// Problem constants (from reference)
#define BB   2
#define SS   1024
#define DD   1024
#define HH   16
#define HD   64
#define NROW 2048   // B*S
#define NW   4      // waves per attention workgroup
#define RCH  (SS / NW)   // 256 r-columns per wave

typedef __bf16 bf16_t;
typedef __attribute__((ext_vector_type(16))) __bf16 v16bf;
typedef __attribute__((ext_vector_type(8)))  __bf16 v8bf;
typedef __attribute__((ext_vector_type(8)))  float  v8f;

union BF16Frag {
  v16bf v;
  v8bf  h[2];
};

__device__ __forceinline__ v8f wmma_bf16(v16bf a, v16bf b, v8f c) {
  // (neg_a, A, neg_b, B, c_mod, C, reuse_a, reuse_b)
  return __builtin_amdgcn_wmma_f32_16x16x32_bf16(false, a, false, b, (short)0, c,
                                                 false, false);
}

// A-matrix 16x32 bf16 fragment (ISA 7.12.2): lane<16 -> M=lane, K in {ko..ko+7, ko+16..ko+23}
// with ko = 0; lane>=16 -> same M range with ko = 8. Two contiguous 8-elem (16B) runs.
__device__ __forceinline__ v16bf load_a(const bf16_t* base16, int ld, int lane) {
  int m  = lane & 15;
  int ko = (lane & 16) ? 8 : 0;
  const bf16_t* p = base16 + (size_t)m * ld + ko;
  BF16Frag f;
  f.h[0] = *(const v8bf*)(p);
  f.h[1] = *(const v8bf*)(p + 16);
  return f.v;
}

// B-matrix 32x16 bf16 fragment: N = lane&15, K in {ko..ko+7, ko+16..ko+23}, ko = (lane&16 ? 8 : 0).
// Two contiguous 8-elem (16B) runs = two b128 loads. Source row index = N, row holds K-contig data.
__device__ __forceinline__ v16bf load_b(const bf16_t* base16, int ld, int lane) {
  int n  = lane & 15;
  int ko = (lane & 16) ? 8 : 0;
  const bf16_t* p = base16 + (size_t)n * ld + ko;
  BF16Frag f;
  f.h[0] = *(const v8bf*)(p);
  f.h[1] = *(const v8bf*)(p + 16);
  return f.v;
}

// ---------------------------------------------------------------------------
// Elementwise f32 -> bf16 conversion
// ---------------------------------------------------------------------------
__global__ void cvt_f32_bf16(const float* __restrict__ src, bf16_t* __restrict__ dst,
                             int n) {
  int i = blockIdx.x * blockDim.x + threadIdx.x;
  if (i < n) dst[i] = (bf16_t)src[i];
}

// ---------------------------------------------------------------------------
// QKV projection: out = hs @ W^T + b.  A = hs[2048x1024] bf16, B = W rows (K-contig).
// Each wave computes TWO 16x16 C tiles (M and M+16) sharing one B fragment:
// 2 WMMAs per 6 b128 loads instead of 1 per 4.
// mode 0/1: write [bh][s][d] (q,k);  mode 2: write transposed [bh][d][s] (v^T).
// Grid (64, 8), block 256 (8 waves).
// ---------------------------------------------------------------------------
__global__ __launch_bounds__(256) void qkv_proj(const bf16_t* __restrict__ hs,
                                                const bf16_t* __restrict__ W,
                                                const float* __restrict__ bias,
                                                bf16_t* __restrict__ outp, int mode) {
  const int lane = threadIdx.x & 31;
  const int wave = threadIdx.x >> 5;
  const int m0 = blockIdx.x * 32;                 // hs row tile pair (token dim)
  const int n0 = (blockIdx.y * 8 + wave) * 16;    // out col tile (feature dim)
  const bf16_t* arow0 = hs + (size_t)m0 * DD;
  const bf16_t* arow1 = hs + (size_t)(m0 + 16) * DD;
  const bf16_t* brow = W + (size_t)n0 * DD;

  v8f accA = {}, accB = {};
#pragma unroll 4
  for (int kk = 0; kk < DD; kk += 32) {
    v16bf b = load_b(brow + kk, DD, lane);
    accA = wmma_bf16(load_a(arow0 + kk, DD, lane), b, accA);
    accB = wmma_bf16(load_a(arow1 + kk, DD, lane), b, accB);
  }

  const float bv = bias[n0 + (lane & 15)];
  const int N = lane & 15;
  const int Mb = (lane & 16) ? 8 : 0;
#pragma unroll
  for (int half = 0; half < 2; ++half) {
#pragma unroll
    for (int i = 0; i < 8; ++i) {
      int nrow = m0 + half * 16 + Mb + i;  // token index in [0,2048)
      int mcol = n0 + N;                   // feature index in [0,1024)
      float vv = (half ? accB[i] : accA[i]) + bv;
      int b_ = nrow >> 10, s = nrow & 1023, h = mcol >> 6, d = mcol & 63;
      size_t bh = (size_t)(b_ * HH + h);
      size_t idx = (mode == 2) ? (bh * HD + (size_t)d) * SS + (size_t)s
                               : (bh * SS + (size_t)s) * HD + (size_t)d;
      outp[idx] = (bf16_t)vv;
    }
  }
}

// ---------------------------------------------------------------------------
// Fused stochastic attention. One workgroup = (b,h, 16 query rows).
// 4 waves split the 1024 r-columns; scores row block resident in LDS (64KB).
// Phase 1: scores = (QK^T + rel)/8 + mask  -> LDS;  online (m1, Z1) of scores+gumbel.
// Phase 2: samples = exp(s1-m1)/Z1; t = scores + samples; flash softmax + WMMA PV.
// ---------------------------------------------------------------------------
__global__ __launch_bounds__(128) void stoch_attn(
    const bf16_t* __restrict__ qbf, const bf16_t* __restrict__ kbf,
    const bf16_t* __restrict__ vT, const bf16_t* __restrict__ ebf,
    const float* __restrict__ mask, const float* __restrict__ gum_u,
    float* __restrict__ out) {
  __shared__ float  sc[16][SS];            // 64 KB score row block
  __shared__ float  mrel[NW][16][32];      // rel-pos GEMM tiles (diagonal gather)
  __shared__ bf16_t pstage[NW][16][32];    // probs staging for PV A-fragments
  __shared__ float  red[NW][16][2];        // per-wave (m,Z) partials
  __shared__ float  accw[NW][16][HD];      // per-wave ctx partials

  const int lane = threadIdx.x & 31;
  const int wave = threadIdx.x >> 5;
  const int l0 = blockIdx.x * 16;
  const int bh = blockIdx.y;  // b*16 + h
  const int b_ = bh >> 4;

  const int N = lane & 15;                 // C-layout column
  const int Mb = (lane & 16) ? 8 : 0;      // C-layout row base
  const int r0w = wave * RCH;

  const bf16_t* qrow  = qbf + ((size_t)bh * SS + (size_t)l0) * HD;
  const bf16_t* kbase = kbf + (size_t)bh * SS * HD;
  const bf16_t* vbase = vT + (size_t)bh * HD * SS;
  const float*  gum   = gum_u + ((size_t)bh * SS + (size_t)l0) * SS;
  const float*  mrow  = mask + (size_t)b_ * SS;

  // Q fragments for this row block (reused by every tile)
  v16bf qa0 = load_a(qrow, HD, lane);
  v16bf qa1 = load_a(qrow + 32, HD, lane);

  float m1p[8], Z1p[8];
#pragma unroll
  for (int i = 0; i < 8; ++i) { m1p[i] = -1e30f; Z1p[i] = 0.f; }

  // ------------------ Phase 1 ------------------
  for (int t = 0; t < RCH / 16; ++t) {
    int r0 = r0w + t * 16;

    // prefetch next r-tile's gumbel lines (global_prefetch_b8)
    if (t + 1 < RCH / 16) {
#pragma unroll
      for (int i = 0; i < 8; ++i)
        __builtin_prefetch(&gum[(size_t)(Mb + i) * SS + (size_t)(r0 + 16 + N)], 0, 3);
    }

    const bf16_t* kr = kbase + (size_t)r0 * HD;
    v8f c = {};
    c = wmma_bf16(qa0, load_b(kr, HD, lane), c);
    c = wmma_bf16(qa1, load_b(kr + 32, HD, lane), c);

    // rel-pos: M[dl, j] = q[l0+dl] . E[jb + j],  rel[dl,dr] = M[dl, dl-dr+15]
    int jb = l0 - r0 + 1008;  // in [0, 2016]
    const bf16_t* er = ebf + (size_t)jb * HD;
    v8f mr0 = {}, mr1 = {};
    mr0 = wmma_bf16(qa0, load_b(er, HD, lane), mr0);
    mr0 = wmma_bf16(qa1, load_b(er + 32, HD, lane), mr0);
    const bf16_t* er2 = er + 16 * HD;
    mr1 = wmma_bf16(qa0, load_b(er2, HD, lane), mr1);
    mr1 = wmma_bf16(qa1, load_b(er2 + 32, HD, lane), mr1);

#pragma unroll
    for (int i = 0; i < 8; ++i) {
      mrel[wave][Mb + i][N] = mr0[i];
      mrel[wave][Mb + i][16 + N] = mr1[i];
    }
    asm volatile("s_wait_dscnt 0" ::: "memory");  // cross-lane LDS RAW

    float mk = mrow[r0 + N];
#pragma unroll
    for (int i = 0; i < 8; ++i) {
      int M = Mb + i;
      float rel = mrel[wave][M][M - N + 15];
      float sv = (c[i] + rel) * 0.125f + mk;
      sc[M][r0 + N] = sv;
      float u = gum[(size_t)M * SS + (size_t)(r0 + N)];
      float g = -__logf(-__logf(u + 1e-10f) + 1e-10f);
      float s1 = sv + g;
      float mn = fmaxf(m1p[i], s1);
      Z1p[i] = Z1p[i] * __expf(m1p[i] - mn) + __expf(s1 - mn);
      m1p[i] = mn;
    }
  }

  // reduce (m1,Z1) across the 16 lanes holding each row, then across waves
#pragma unroll
  for (int i = 0; i < 8; ++i) {
#pragma unroll
    for (int d = 1; d < 16; d <<= 1) {
      float mo = __shfl_xor(m1p[i], d, 32);
      float Zo = __shfl_xor(Z1p[i], d, 32);
      float mn = fmaxf(m1p[i], mo);
      Z1p[i] = Z1p[i] * __expf(m1p[i] - mn) + Zo * __expf(mo - mn);
      m1p[i] = mn;
    }
    if (N == 0) { red[wave][Mb + i][0] = m1p[i]; red[wave][Mb + i][1] = Z1p[i]; }
  }
  __syncthreads();

  float m1f[8], iZ1[8];
#pragma unroll
  for (int i = 0; i < 8; ++i) {
    float mA = -1e30f, ZA = 0.f;
#pragma unroll
    for (int w = 0; w < NW; ++w) {
      float mw = red[w][Mb + i][0], Zw = red[w][Mb + i][1];
      float mn = fmaxf(mA, mw);
      ZA = ZA * __expf(mA - mn) + Zw * __expf(mw - mn);
      mA = mn;
    }
    m1f[i] = mA;
    iZ1[i] = 1.0f / ZA;
  }
  __syncthreads();  // all waves consumed red before phase 2 reuses it

  // ------------------ Phase 2 ------------------
  float m2p[8], Z2p[8];
  v8f acc0 = {}, acc1 = {}, acc2 = {}, acc3 = {};
#pragma unroll
  for (int i = 0; i < 8; ++i) { m2p[i] = -1e30f; Z2p[i] = 0.f; }

  for (int t2 = 0; t2 < RCH / 32; ++t2) {
    int rb = r0w + t2 * 32;

    if (t2 + 1 < RCH / 32) {
#pragma unroll
      for (int i = 0; i < 8; ++i)
        __builtin_prefetch(&gum[(size_t)(Mb + i) * SS + (size_t)(rb + 32 + N)], 0, 3);
    }

    float tv0[8], tv1[8];
#pragma unroll
    for (int i = 0; i < 8; ++i) {
      int M = Mb + i;
      {
        float sv = sc[M][rb + N];
        float u = gum[(size_t)M * SS + (size_t)(rb + N)];
        float g = -__logf(-__logf(u + 1e-10f) + 1e-10f);
        float samp = __expf(sv + g - m1f[i]) * iZ1[i];
        tv0[i] = sv + samp;  // TAU_1 == 1
      }
      {
        float sv = sc[M][rb + 16 + N];
        float u = gum[(size_t)M * SS + (size_t)(rb + 16 + N)];
        float g = -__logf(-__logf(u + 1e-10f) + 1e-10f);
        float samp = __expf(sv + g - m1f[i]) * iZ1[i];
        tv1[i] = sv + samp;
      }
    }
    float rsc[8];
#pragma unroll
    for (int i = 0; i < 8; ++i) {
      float lm = fmaxf(tv0[i], tv1[i]);
#pragma unroll
      for (int d = 1; d < 16; d <<= 1) lm = fmaxf(lm, __shfl_xor(lm, d, 32));
      float mn = fmaxf(m2p[i], lm);
      rsc[i] = __expf(m2p[i] - mn);
      float p0 = __expf(tv0[i] - mn);
      float p1 = __expf(tv1[i] - mn);
      Z2p[i] = Z2p[i] * rsc[i] + p0 + p1;
      m2p[i] = mn;
      pstage[wave][Mb + i][N] = (bf16_t)p0;
      pstage[wave][Mb + i][16 + N] = (bf16_t)p1;
    }
    asm volatile("s_wait_dscnt 0" ::: "memory");

#pragma unroll
    for (int i = 0; i < 8; ++i) {  // flash rescale of accumulators (row-uniform)
      acc0[i] = acc0[i] * rsc[i];
      acc1[i] = acc1[i] * rsc[i];
      acc2[i] = acc2[i] * rsc[i];
      acc3[i] = acc3[i] * rsc[i];
    }
    v16bf pa = load_a(&pstage[wave][0][0], 32, lane);
    acc0 = wmma_bf16(pa, load_b(vbase + (size_t)(0 * 16) * SS + rb, SS, lane), acc0);
    acc1 = wmma_bf16(pa, load_b(vbase + (size_t)(1 * 16) * SS + rb, SS, lane), acc1);
    acc2 = wmma_bf16(pa, load_b(vbase + (size_t)(2 * 16) * SS + rb, SS, lane), acc2);
    acc3 = wmma_bf16(pa, load_b(vbase + (size_t)(3 * 16) * SS + rb, SS, lane), acc3);
  }

  // per-wave partial stats + context into LDS
#pragma unroll
  for (int i = 0; i < 8; ++i) {
    float z = Z2p[i];
#pragma unroll
    for (int d = 1; d < 16; d <<= 1) z += __shfl_xor(z, d, 32);
    if (N == 0) { red[wave][Mb + i][0] = m2p[i]; red[wave][Mb + i][1] = z; }
    accw[wave][Mb + i][0 + N] = acc0[i];
    accw[wave][Mb + i][16 + N] = acc1[i];
    accw[wave][Mb + i][32 + N] = acc2[i];
    accw[wave][Mb + i][48 + N] = acc3[i];
  }
  __syncthreads();

  // merge 4 wave partials, normalize, write out[b, s, h*64+d]
  const int tid = threadIdx.x;
  const int row = tid >> 3;          // 0..15
  const int dbase = (tid & 7) * 8;   // 0..56
  float mf = -1e30f, Zf = 0.f;
  for (int w = 0; w < NW; ++w) {
    float mw = red[w][row][0], Zw = red[w][row][1];
    float mn = fmaxf(mf, mw);
    Zf = Zf * __expf(mf - mn) + Zw * __expf(mw - mn);
    mf = mn;
  }
  float fw[NW];
  for (int w = 0; w < NW; ++w) fw[w] = __expf(red[w][row][0] - mf);
  float iZ = 1.0f / Zf;
  float* op = out + ((size_t)b_ * SS + (size_t)(l0 + row)) * DD + (size_t)(bh & 15) * HD;
#pragma unroll
  for (int e = 0; e < 8; ++e) {
    int dd = dbase + e;
    float v = 0.f;
    for (int w = 0; w < NW; ++w) v += accw[w][row][dd] * fw[w];
    op[dd] = v * iZ;
  }
}

// ---------------------------------------------------------------------------
extern "C" void kernel_launch(void* const* d_in, const int* in_sizes, int n_in,
                              void* d_out, int out_size, void* d_ws, size_t ws_size,
                              hipStream_t stream) {
  (void)in_sizes; (void)n_in; (void)out_size; (void)ws_size;
  const float* hs   = (const float*)d_in[0];
  const float* mask = (const float*)d_in[1];
  const float* gum  = (const float*)d_in[2];
  const float* Wq   = (const float*)d_in[3];
  const float* bq   = (const float*)d_in[4];
  const float* Wk   = (const float*)d_in[5];
  const float* bk   = (const float*)d_in[6];
  const float* Wv   = (const float*)d_in[7];
  const float* bv   = (const float*)d_in[8];
  const float* de   = (const float*)d_in[9];
  float* out = (float*)d_out;

  char* ws = (char*)d_ws;
  bf16_t* hsb = (bf16_t*)(ws);                       // 2048x1024 bf16 (4 MB)
  bf16_t* wqb = (bf16_t*)(ws + ((size_t)4 << 20));   // 1024x1024 (2 MB)
  bf16_t* wkb = (bf16_t*)(ws + ((size_t)6 << 20));
  bf16_t* wvb = (bf16_t*)(ws + ((size_t)8 << 20));
  bf16_t* qb  = (bf16_t*)(ws + ((size_t)10 << 20));  // [bh][s][d] (4 MB)
  bf16_t* kb  = (bf16_t*)(ws + ((size_t)14 << 20));  // [bh][s][d]
  bf16_t* vtb = (bf16_t*)(ws + ((size_t)18 << 20));  // [bh][d][s] transposed
  bf16_t* eb  = (bf16_t*)(ws + ((size_t)22 << 20));  // 2048x64 (padded row 2047 unused)

  cvt_f32_bf16<<<(NROW * DD + 255) / 256, 256, 0, stream>>>(hs, hsb, NROW * DD);
  cvt_f32_bf16<<<(DD * DD + 255) / 256, 256, 0, stream>>>(Wq, wqb, DD * DD);
  cvt_f32_bf16<<<(DD * DD + 255) / 256, 256, 0, stream>>>(Wk, wkb, DD * DD);
  cvt_f32_bf16<<<(DD * DD + 255) / 256, 256, 0, stream>>>(Wv, wvb, DD * DD);
  cvt_f32_bf16<<<(2047 * HD + 255) / 256, 256, 0, stream>>>(de, eb, 2047 * HD);

  dim3 pg(NROW / 32, 64 / 8, 1);
  qkv_proj<<<pg, 256, 0, stream>>>(hsb, wqb, bq, qb, 0);
  qkv_proj<<<pg, 256, 0, stream>>>(hsb, wkb, bk, kb, 1);
  qkv_proj<<<pg, 256, 0, stream>>>(hsb, wvb, bv, vtb, 2);

  stoch_attn<<<dim3(SS / 16, BB * HH), 128, 0, stream>>>(qb, kb, vtb, eb, mask, gum,
                                                         out);
}
